// ODE_RNN_4389456577078
// MI455X (gfx1250) — compile-verified, matrix operand order checked by reading, not run on hardware
//
#include <hip/hip_runtime.h>
#include <math.h>

#define T_LEN   32768
#define IN_DIM  1024
#define HID     512
#define OUT_DIM 128

typedef __attribute__((ext_vector_type(16))) _Float16 v16h;
typedef __attribute__((ext_vector_type(8)))  _Float16 v8h;
typedef __attribute__((ext_vector_type(8)))  float    v8f;

// ---------------------------------------------------------------------------
// Kernel 1: pack W_ode and W_h into a [k/4][r][k%4] float layout so the scan
// kernel's per-lane float4 load (lane r -> 16B at (k4*512 + r)) is a fully
// coalesced 512B wave request.
// ---------------------------------------------------------------------------
__global__ __launch_bounds__(256) void pack_w_kernel(const float* __restrict__ Wode,
                                                     const float* __restrict__ Wh,
                                                     float* __restrict__ ode_p,
                                                     float* __restrict__ wh_p) {
  int idx = blockIdx.x * 256 + threadIdx.x;       // 0 .. 2*512*512-1 (exact)
  const float* src;
  float* dst;
  int e = idx;
  if (idx < HID * HID) { src = Wode; dst = ode_p; }
  else                 { src = Wh;   dst = wh_p;  e -= HID * HID; }
  int r = e >> 9;          // row    (output index)
  int k = e & 511;         // column (reduction index)
  dst[(k >> 2) * (HID * 4) + r * 4 + (k & 3)] = src[e];
}

// ---------------------------------------------------------------------------
// Kernel 2: one-time f32 -> f16 conversion of W_in (re-read by every block of
// the GEMM, so converting once removes per-fragment cvt chains and halves the
// B-operand vmem count).
// ---------------------------------------------------------------------------
__global__ __launch_bounds__(256) void cvt_win_kernel(const float* __restrict__ Win,
                                                      _Float16* __restrict__ Winh) {
  int idx = blockIdx.x * 256 + threadIdx.x;       // 0 .. 512*1024-1 (exact)
  Winh[idx] = (_Float16)Win[idx];
}

// ---------------------------------------------------------------------------
// Kernel 3: U[i][j] = sum_c x[i][c] * W_in[j][c] + b_in[j] + b_h[j]
// via v_wmma_f32_16x16x32_f16.  Per-lane operand fragments follow the CDNA5
// 16-bit layout: lane = (row&15) + 16*khalf; halves 0..7 = K[c..c+7],
// halves 8..15 = K[c+16..c+23], with c = k0 + 8*khalf.
// 4 waves per block; each wave owns 128 output columns (8 j-tiles) so the
// A fragment is reused across 8 WMMAs.
// ---------------------------------------------------------------------------
__device__ __forceinline__ v16h load_frag_f32_to_f16(const float* __restrict__ row, int c) {
  float4 a0 = *(const float4*)(row + c);
  float4 a1 = *(const float4*)(row + c + 4);
  float4 a2 = *(const float4*)(row + c + 16);
  float4 a3 = *(const float4*)(row + c + 20);
  v16h A;
  A[0]  = (_Float16)a0.x; A[1]  = (_Float16)a0.y; A[2]  = (_Float16)a0.z; A[3]  = (_Float16)a0.w;
  A[4]  = (_Float16)a1.x; A[5]  = (_Float16)a1.y; A[6]  = (_Float16)a1.z; A[7]  = (_Float16)a1.w;
  A[8]  = (_Float16)a2.x; A[9]  = (_Float16)a2.y; A[10] = (_Float16)a2.z; A[11] = (_Float16)a2.w;
  A[12] = (_Float16)a3.x; A[13] = (_Float16)a3.y; A[14] = (_Float16)a3.z; A[15] = (_Float16)a3.w;
  return A;
}

__device__ __forceinline__ v16h load_frag_f16(const _Float16* __restrict__ row, int c) {
  v8h lo = *(const v8h*)(row + c);        // halves 0..7  = K[c .. c+7]
  v8h hi = *(const v8h*)(row + c + 16);   // halves 8..15 = K[c+16 .. c+23]
  return __builtin_shufflevector(lo, hi, 0, 1, 2, 3, 4, 5, 6, 7,
                                         8, 9, 10, 11, 12, 13, 14, 15);
}

__global__ __launch_bounds__(128) void gemm_u_kernel(const float* __restrict__ X,
                                                     const _Float16* __restrict__ Winh,
                                                     const float* __restrict__ b_in,
                                                     const float* __restrict__ b_h,
                                                     float* __restrict__ U) {
  const int i0    = blockIdx.x * 16;       // 16 sequence rows per block
  const int wave  = threadIdx.x >> 5;      // 4 waves; wave owns 128 output cols
  const int lane  = threadIdx.x & 31;
  const int m     = lane & 15;
  const int khalf = lane >> 4;
  const int j0    = wave * 128;

  const float* xrow = X + (size_t)(i0 + m) * IN_DIM;
  const _Float16* wrow[8];
#pragma unroll
  for (int jt = 0; jt < 8; ++jt)
    wrow[jt] = Winh + (size_t)(j0 + jt * 16 + m) * IN_DIM;

  v8f acc[8] = {};

  for (int k0 = 0; k0 < IN_DIM; k0 += 32) {
    const int c = k0 + 8 * khalf;
    v16h A = load_frag_f32_to_f16(xrow, c);
#pragma unroll
    for (int jt = 0; jt < 8; ++jt) {
      v16h B = load_frag_f16(wrow[jt], c);
      acc[jt] = __builtin_amdgcn_wmma_f32_16x16x32_f16(
          /*neg_a=*/false, A, /*neg_b=*/false, B,
          /*c_mod=*/(short)0, acc[jt], /*reuse_a=*/false, /*reuse_b=*/false);
    }
  }

#pragma unroll
  for (int jt = 0; jt < 8; ++jt) {
    const int j = j0 + jt * 16 + m;                 // N index = lane&15
    const float bias = b_in[j] + b_h[j];
#pragma unroll
    for (int v = 0; v < 8; ++v) {
      const int i = i0 + v + 8 * khalf;             // M index = v + 8*khalf
      U[(size_t)i * HID + j] = acc[jt][v] + bias;
    }
  }
}

// ---------------------------------------------------------------------------
// Kernel 4: persistent single-workgroup sequential scan.
// thread r owns h[r]; weights read (coalesced float4) from L2-resident packed
// buffers; stage vectors exchanged through LDS with workgroup barriers.
// ---------------------------------------------------------------------------
__device__ __forceinline__ float dot_packed(const float* __restrict__ Wp,
                                            const float* sv, int r) {
  const float4* w = (const float4*)Wp;
  float a0 = 0.f, a1 = 0.f, a2 = 0.f, a3 = 0.f;
#pragma unroll 4
  for (int k4 = 0; k4 < HID / 4; k4 += 4) {
    float4 w0 = w[(k4 + 0) * HID + r];
    float4 w1 = w[(k4 + 1) * HID + r];
    float4 w2 = w[(k4 + 2) * HID + r];
    float4 w3 = w[(k4 + 3) * HID + r];
    a0 += w0.x * sv[4*k4 +  0] + w0.y * sv[4*k4 +  1] + w0.z * sv[4*k4 +  2] + w0.w * sv[4*k4 +  3];
    a1 += w1.x * sv[4*k4 +  4] + w1.y * sv[4*k4 +  5] + w1.z * sv[4*k4 +  6] + w1.w * sv[4*k4 +  7];
    a2 += w2.x * sv[4*k4 +  8] + w2.y * sv[4*k4 +  9] + w2.z * sv[4*k4 + 10] + w2.w * sv[4*k4 + 11];
    a3 += w3.x * sv[4*k4 + 12] + w3.y * sv[4*k4 + 13] + w3.z * sv[4*k4 + 14] + w3.w * sv[4*k4 + 15];
  }
  return (a0 + a1) + (a2 + a3);
}

__global__ __launch_bounds__(512) void scan_kernel(const float* __restrict__ t,
                                                   const float* __restrict__ U,
                                                   const float* __restrict__ ode_p,
                                                   const float* __restrict__ wh_p,
                                                   const float* __restrict__ b_ode,
                                                   const float* __restrict__ Wdec,
                                                   const float* __restrict__ b_dec,
                                                   float* __restrict__ out) {
  __shared__ float h[HID];
  __shared__ float sv[HID];
  const int r = threadIdx.x;
  const float bo = b_ode[r];

  // i == 0: h0 = tanh(W_in x_0 + b_in + b_h) = tanh(U[0])
  h[r] = tanhf(U[r]);
  __syncthreads();

#pragma unroll 1
  for (int i = 1; i < T_LEN; ++i) {
    const float dt_total = t[i] - t[i - 1];
    const float dt  = 0.25f * dt_total;    // N_SUB = 4 substeps
    const float hdt = 0.5f * dt;

#pragma unroll 1
    for (int ss = 0; ss < 4; ++ss) {
      const float h_r = h[r];
      const float k1 = tanhf(bo + dot_packed(ode_p, h, r));
      sv[r] = h_r + hdt * k1;
      __syncthreads();
      const float k2 = tanhf(bo + dot_packed(ode_p, sv, r));
      __syncthreads();
      sv[r] = h_r + hdt * k2;
      __syncthreads();
      const float k3 = tanhf(bo + dot_packed(ode_p, sv, r));
      __syncthreads();
      sv[r] = h_r + dt * k3;
      __syncthreads();
      const float k4 = tanhf(bo + dot_packed(ode_p, sv, r));
      __syncthreads();                       // everyone done reading sv and h
      h[r] = h_r + (dt * (1.0f / 6.0f)) * (k1 + 2.0f * k2 + 2.0f * k3 + k4);
      __syncthreads();
    }

    // h_new = tanh(U[i] + W_h @ h_i)   (U already contains b_in + b_h)
    const float pre = U[(size_t)i * HID + r] + dot_packed(wh_p, h, r);
    __syncthreads();                         // everyone done reading h
    h[r] = tanhf(pre);
    __syncthreads();
  }

  // decode: out = W_dec @ h + b_dec
  if (r < OUT_DIM) {
    float acc = b_dec[r];
    for (int k = 0; k < HID; ++k) acc += Wdec[r * HID + k] * h[k];
    out[r] = acc;
  }
}

// ---------------------------------------------------------------------------
// Launch
// ---------------------------------------------------------------------------
extern "C" void kernel_launch(void* const* d_in, const int* in_sizes, int n_in,
                              void* d_out, int out_size, void* d_ws, size_t ws_size,
                              hipStream_t stream) {
  (void)in_sizes; (void)n_in; (void)out_size; (void)ws_size;
  const float* t_arr = (const float*)d_in[0];
  const float* x     = (const float*)d_in[1];
  const float* W_in  = (const float*)d_in[2];
  const float* b_in  = (const float*)d_in[3];
  const float* W_h   = (const float*)d_in[4];
  const float* b_h   = (const float*)d_in[5];
  const float* W_ode = (const float*)d_in[6];
  const float* b_ode = (const float*)d_in[7];
  const float* W_dec = (const float*)d_in[8];
  const float* b_dec = (const float*)d_in[9];

  float*    ws    = (float*)d_ws;
  float*    ode_p = ws;                          // 512*512 floats (1 MB)
  float*    wh_p  = ws + HID * HID;              // 512*512 floats (1 MB)
  float*    U     = ws + 2 * HID * HID;          // 32768*512 floats (64 MB)
  _Float16* Winh  = (_Float16*)(ws + 2 * HID * HID + (size_t)T_LEN * HID); // 512*1024 f16 (1 MB)

  // 1) pack scan weights for coalesced matvec access
  pack_w_kernel<<<(2 * HID * HID) / 256, 256, 0, stream>>>(W_ode, W_h, ode_p, wh_p);

  // 2) one-time f16 conversion of W_in
  cvt_win_kernel<<<(HID * IN_DIM) / 256, 256, 0, stream>>>(W_in, Winh);

  // 3) WMMA GEMM: U = X @ W_in^T + (b_in + b_h)
  gemm_u_kernel<<<T_LEN / 16, 128, 0, stream>>>(x, Winh, b_in, b_h, U);

  // 4) sequential ODE-RNN scan (single persistent workgroup) + decode
  scan_kernel<<<1, HID, 0, stream>>>(t_arr, U, ode_p, wh_p, b_ode, W_dec, b_dec,
                                     (float*)d_out);
}
